// NKIMoELayer_77670188581355
// MI455X (gfx1250) — compile-verified, matrix operand order checked by reading, not run on hardware
//
#include <hip/hip_runtime.h>
#include <hip/hip_bf16.h>

// ---------------------------------------------------------------------------
// Problem constants (from reference): B=4,S=1024,H=2048, E=16,I=2816,K=8
// ---------------------------------------------------------------------------
constexpr int B_ = 4, S_ = 1024, H_ = 2048;
constexpr int E_ = 16, I_ = 2816, K_ = 8;
constexpr int T_  = B_ * S_;        // 4096 tokens
constexpr int TK_ = T_ * K_;        // 32768 routed (token,expert) entries
constexpr int TWO_I = 2 * I_;       // 5632

constexpr int MT = 32;              // rows per tile
constexpr int NT = 256;             // cols per tile
constexpr int KT = 32;              // k per bf16-WMMA chunk
constexpr int KS = 352;             // k-slab for TDM A panel in GEMM2 (11 chunks)
constexpr int MAX_TILES = TK_ / MT + E_;   // worst-case total M-tiles = 1040

// GEMM2 dynamic LDS layout: A slab at offset 0 (TDM target), then B slab
constexpr unsigned A2_BYTES = MT * KS * 2;            // 22528
constexpr unsigned B2_BYTES = NT * KT * 2;            // 16384
constexpr unsigned G2_LDS   = A2_BYTES + B2_BYTES;    // 38912

// ---------------------------------------------------------------------------
// Workspace layout (bytes)
// ---------------------------------------------------------------------------
constexpr size_t OFF_CNT  = 0;                                   // E ints
constexpr size_t OFF_NT   = 256;                                 // 1 int
constexpr size_t OFF_BASE = 512;                                 // E ints
constexpr size_t OFF_TE   = 1024;                                // MAX_TILES ints
constexpr size_t OFF_TR   = OFF_TE + 4352;                       // MAX_TILES ints
constexpr size_t OFF_TOK  = 16384;                               // E*TK ints (2 MB)
constexpr size_t OFF_W    = OFF_TOK + (size_t)E_ * TK_ * 4;      // E*TK floats (2 MB)
constexpr size_t OFF_ACT  = OFF_W + (size_t)E_ * TK_ * 4;        // TK x I bf16 (184 MB)
constexpr size_t WS_NEEDED = OFF_ACT + (size_t)TK_ * I_ * 2;

// ---------------------------------------------------------------------------
// Types / helpers
// ---------------------------------------------------------------------------
typedef __attribute__((ext_vector_type(16))) __bf16 v16bf;
typedef __attribute__((ext_vector_type(8)))  __bf16 v8bf;
typedef __attribute__((ext_vector_type(2)))  __bf16 v2bf;
typedef __attribute__((ext_vector_type(8)))  float  v8f;
typedef __attribute__((ext_vector_type(4)))  unsigned int v4u;
typedef __attribute__((ext_vector_type(8)))  int v8i;
typedef __attribute__((ext_vector_type(4)))  int v4i;

#if __has_builtin(__builtin_amdgcn_tensor_load_to_lds) && \
    __has_builtin(__builtin_amdgcn_s_wait_tensorcnt)
#define HAVE_TDM 1
#pragma message("CDNA5 probe: TDM=1 (tensor_load_to_lds path active in this pass)")
#else
#define HAVE_TDM 0
#pragma message("CDNA5 probe: TDM=0 (fallback copy path in this pass)")
#endif

__device__ __forceinline__ __bf16 f2bf(float f) {
    unsigned u = __builtin_bit_cast(unsigned, f);
    u += 0x7fffu + ((u >> 16) & 1u);                 // round-to-nearest-even
    unsigned short h = (unsigned short)(u >> 16);
    return __builtin_bit_cast(__bf16, h);
}

#if __has_builtin(__builtin_amdgcn_cvt_pk_bf16_f32)
#pragma message("CDNA5 probe: PKCVT=1 (v_cvt_pk_bf16_f32 active in this pass)")
__device__ __forceinline__ v2bf f2bf2(float a, float b) {
    return __builtin_amdgcn_cvt_pk_bf16_f32(a, b);
}
#else
#pragma message("CDNA5 probe: PKCVT=0 (software bf16 rounding in this pass)")
__device__ __forceinline__ v2bf f2bf2(float a, float b) {
    v2bf r; r[0] = f2bf(a); r[1] = f2bf(b); return r;
}
#endif

// Load one 16-lane-striped bf16 WMMA fragment: two contiguous 16B chunks.
__device__ __forceinline__ v16bf load_frag(const __bf16* row) {
    v8bf lo = *(const v8bf*)(row);
    v8bf hi = *(const v8bf*)(row + 16);
    return __builtin_shufflevector(lo, hi, 0, 1, 2, 3, 4, 5, 6, 7,
                                           8, 9, 10, 11, 12, 13, 14, 15);
}

#if HAVE_TDM
// Issue a 2D TDM tile load: tile_d0 x tile_d1 elements (data_size 2^dsz_log
// bytes each, row stride stride0 elements) from global -> LDS byte offset
// lds_off. Descriptor per CDNA5 D# spec (count=1, type=2 "image").
__device__ __forceinline__ void tdm_load_2d(
    unsigned lds_off, const void* gptr,
    unsigned tile_d0, unsigned tile_d1,
    unsigned tensor_d0, unsigned tensor_d1,
    unsigned stride0, unsigned dsz_log)
{
    unsigned long long ga = (unsigned long long)(size_t)gptr;
    v4u g0;
    g0[0] = 1u;                                            // count=1, user mode
    g0[1] = lds_off;                                       // lds_addr
    g0[2] = (unsigned)(ga & 0xFFFFFFFFu);                  // global_addr[31:0]
    g0[3] = (unsigned)((ga >> 32) & 0x01FFFFFFu) | (2u << 30);  // [56:32] | type=2
    v8i g1;
    g1[0] = (int)(dsz_log << 16);                          // wg_mask=0 | data_size
    g1[1] = (int)((tensor_d0 & 0xFFFFu) << 16);            // tensor_dim0[15:0]
    g1[2] = (int)((tensor_d0 >> 16) | ((tensor_d1 & 0xFFFFu) << 16));
    g1[3] = (int)((tensor_d1 >> 16) | (tile_d0 << 16));    // | tile_dim0
    g1[4] = (int)tile_d1;                                  // tile_dim1 | tile_dim2=0
    g1[5] = (int)stride0;                                  // tensor_dim0_stride[31:0]
    g1[6] = 0;                                             // stride0[47:32]|stride1 lo
    g1[7] = 0;
    v4i z4 = {0, 0, 0, 0};
#if __clang_major__ >= 23
    v8i z8 = {0, 0, 0, 0, 0, 0, 0, 0};
    __builtin_amdgcn_tensor_load_to_lds(g0, g1, z4, z4, z8, 0);
#else
    __builtin_amdgcn_tensor_load_to_lds(g0, g1, z4, z4, 0);
#endif
}
#endif

// ---------------------------------------------------------------------------
// Routing
// ---------------------------------------------------------------------------
__global__ void k_init(int* cnt, int* nTiles) {
    if (threadIdx.x < E_) cnt[threadIdx.x] = 0;
    if (threadIdx.x == 0) *nTiles = 0;
}

__global__ void k_route(const int* __restrict__ idx, const float* __restrict__ wts,
                        int* cnt, int* listTok, float* listW) {
    int t = blockIdx.x * blockDim.x + threadIdx.x;
    if (t >= T_) return;
    #pragma unroll
    for (int k = 0; k < K_; ++k) {
        int e   = idx[t * K_ + k] & (E_ - 1);
        float w = wts[t * K_ + k];
        int pos = atomicAdd(&cnt[e], 1);
        listTok[e * TK_ + pos] = t;
        listW[e * TK_ + pos]   = w;
    }
}

__global__ void k_tiles(const int* cnt, int* baseE, int* tileE, int* tileR, int* nTiles) {
    if (threadIdx.x != 0 || blockIdx.x != 0) return;
    int total = 0, base = 0;
    for (int e = 0; e < E_; ++e) {
        baseE[e] = base;
        int c = cnt[e];
        base += c;
        int nt = (c + MT - 1) / MT;
        for (int i = 0; i < nt; ++i) { tileE[total] = e; tileR[total] = i * MT; ++total; }
    }
    *nTiles = total;
}

__global__ void k_zero(float* out) {
    size_t i = (size_t)blockIdx.x * blockDim.x + threadIdx.x;
    ((float4*)out)[i] = make_float4(0.f, 0.f, 0.f, 0.f);
}

// ---------------------------------------------------------------------------
// GEMM1: act[row, n] = silu(x_row . Wg[:,n]) * (x_row . Wu[:,n])   (bf16 out)
// Block: 32 rows x 256 cols, 8 waves (each wave a 32x32 g-subtile + u-subtile)
// ---------------------------------------------------------------------------
__global__ __launch_bounds__(256) void k_gemm1(
    const float* __restrict__ x, const float* __restrict__ gup,
    const int* __restrict__ tileE, const int* __restrict__ tileR,
    const int* __restrict__ nTiles, const int* __restrict__ cnt,
    const int* __restrict__ baseE, const int* __restrict__ listTok,
    __bf16* __restrict__ act)
{
    __shared__ __align__(16) __bf16 a_lds[MT * KT];       // 32x32, m-major
    __shared__ __align__(16) __bf16 bg_lds[NT * KT];      // 256x32, n-major
    __shared__ __align__(16) __bf16 bu_lds[NT * KT];

    const int tile = blockIdx.x;
    if (tile >= *nTiles) return;
    const int e       = tileE[tile];
    const int r0      = tileR[tile];
    const int rows    = min(MT, cnt[e] - r0);
    const int actRow0 = baseE[e] + r0;
    const int nBase   = blockIdx.y * NT;

    const int tid = threadIdx.x;
    const int m  = tid >> 3;
    const int k4 = (tid & 7) << 2;
    const int tok = (m < rows) ? listTok[e * TK_ + r0 + m] : -1;
    const float* xrow = x + (size_t)(tok < 0 ? 0 : tok) * H_;

    const float* gcol = gup + (size_t)e * H_ * TWO_I + nBase + tid;
    const float* ucol = gcol + I_;

    const int w = tid >> 5, lane = tid & 31;
    const int half = lane >> 4, lrow = lane & 15;

    v8f gacc[2][2] = {{{0}}};
    v8f uacc[2][2] = {{{0}}};

    for (int kb = 0; kb < H_; kb += KT) {
        // ---- stage A tile (f32 -> bf16, packed converts) ----
        float4 av = make_float4(0.f, 0.f, 0.f, 0.f);
        if (tok >= 0) av = *(const float4*)(xrow + kb + k4);
        *(v2bf*)&a_lds[m * KT + k4 + 0] = f2bf2(av.x, av.y);
        *(v2bf*)&a_lds[m * KT + k4 + 2] = f2bf2(av.z, av.w);

        // ---- stage B slabs (g and u), transposed to n-major ----
        const float* gp = gcol + (size_t)kb * TWO_I;
        const float* up = ucol + (size_t)kb * TWO_I;
        #pragma unroll
        for (int k8 = 0; k8 < KT; k8 += 8) {
            v8bf pg, pu;
            #pragma unroll
            for (int j = 0; j < 8; j += 2) {
                v2bf tg = f2bf2(gp[(size_t)(k8 + j) * TWO_I],
                                gp[(size_t)(k8 + j + 1) * TWO_I]);
                v2bf tu = f2bf2(up[(size_t)(k8 + j) * TWO_I],
                                up[(size_t)(k8 + j + 1) * TWO_I]);
                pg[j] = tg[0]; pg[j + 1] = tg[1];
                pu[j] = tu[0]; pu[j + 1] = tu[1];
            }
            *(v8bf*)&bg_lds[tid * KT + k8] = pg;
            *(v8bf*)&bu_lds[tid * KT + k8] = pu;
        }
        if (kb + KT < H_) {
            __builtin_prefetch(gp + (size_t)KT * TWO_I, 0, 1);
            __builtin_prefetch(up + (size_t)KT * TWO_I, 0, 1);
        }
        __syncthreads();

        // ---- WMMA ----
        v16bf afr[2];
        #pragma unroll
        for (int mi = 0; mi < 2; ++mi)
            afr[mi] = load_frag(&a_lds[(mi * 16 + lrow) * KT + half * 8]);
        #pragma unroll
        for (int ni = 0; ni < 2; ++ni) {
            const int n = w * 32 + ni * 16 + lrow;
            v16bf bg = load_frag(&bg_lds[n * KT + half * 8]);
            v16bf bu = load_frag(&bu_lds[n * KT + half * 8]);
            #pragma unroll
            for (int mi = 0; mi < 2; ++mi) {
                gacc[mi][ni] = __builtin_amdgcn_wmma_f32_16x16x32_bf16(
                    false, afr[mi], false, bg, (short)0, gacc[mi][ni], false, false);
                uacc[mi][ni] = __builtin_amdgcn_wmma_f32_16x16x32_bf16(
                    false, afr[mi], false, bu, (short)0, uacc[mi][ni], false, false);
            }
        }
        __syncthreads();
    }

    // ---- epilogue: silu(g)*u -> bf16 act scratch ----
    #pragma unroll
    for (int mi = 0; mi < 2; ++mi)
        #pragma unroll
        for (int ni = 0; ni < 2; ++ni)
            #pragma unroll
            for (int v = 0; v < 8; ++v) {
                const int ml = mi * 16 + half * 8 + v;       // C layout: M = v + 8*half
                if (ml < rows) {
                    const int nl = w * 32 + ni * 16 + lrow;
                    const float g = gacc[mi][ni][v];
                    const float u = uacc[mi][ni][v];
                    const float a = (g / (1.f + __expf(-g))) * u;
                    act[(size_t)(actRow0 + ml) * I_ + nBase + nl] = f2bf(a);
                }
            }
}

// ---------------------------------------------------------------------------
// GEMM2: out[token, n] += w_entry * (act_row . Wd[:, n])
// A panel (32 x KS bf16, contiguous strided rows) is pulled into LDS by the
// Tensor Data Mover; B slab staged/converted by the waves as before.
// Dynamic LDS: A slab at offset 0 (deterministic TDM lds_addr), then B slab.
// ---------------------------------------------------------------------------
__global__ __launch_bounds__(256) void k_gemm2(
    const __bf16* __restrict__ act, const float* __restrict__ dwn,
    const int* __restrict__ tileE, const int* __restrict__ tileR,
    const int* __restrict__ nTiles, const int* __restrict__ cnt,
    const int* __restrict__ baseE, const int* __restrict__ listTok,
    const float* __restrict__ listW, float* __restrict__ out)
{
    extern __shared__ char smem[];
    __bf16* a_slab = (__bf16*)smem;                 // MT x KS, offset 0
    __bf16* b_lds  = (__bf16*)(smem + A2_BYTES);    // NT x KT, n-major

    const int tile = blockIdx.x;
    if (tile >= *nTiles) return;
    const int e       = tileE[tile];
    const int r0      = tileR[tile];
    const int rows    = min(MT, cnt[e] - r0);
    const int actRow0 = baseE[e] + r0;
    const int nBase   = blockIdx.y * NT;

    const int tid = threadIdx.x;
    const float* bcol = dwn + (size_t)e * I_ * H_ + nBase + tid;

    const int w = tid >> 5, lane = tid & 31;
    const int half = lane >> 4, lrow = lane & 15;

    const unsigned rowsAvail = (unsigned)(TK_ - actRow0);   // clamp TDM reads

    v8f acc[2][2] = {{{0}}};

    for (int ks = 0; ks < I_; ks += KS) {
        // ---- A panel: async TDM load (wave 0 issues; EXEC ignored by TDM) ----
#if HAVE_TDM
        if (w == 0) {
            tdm_load_2d(/*lds_off=*/0u,
                        /*gptr=*/act + (size_t)actRow0 * I_ + ks,
                        /*tile_d0=*/KS, /*tile_d1=*/MT,
                        /*tensor_d0=*/(unsigned)(I_ - ks),
                        /*tensor_d1=*/rowsAvail,
                        /*stride0=*/(unsigned)I_, /*dsz_log=*/1u);
            __builtin_amdgcn_s_wait_tensorcnt(0);
        }
#else
        for (int i = tid; i < MT * KS / 4; i += 256) {
            const int row = (i * 4) / KS, col = (i * 4) % KS;
            uint2 v = make_uint2(0u, 0u);
            if ((unsigned)row < rowsAvail)
                v = *(const uint2*)(act + (size_t)(actRow0 + row) * I_ + ks + col);
            *(uint2*)&a_slab[row * KS + col] = v;
        }
#endif
        __syncthreads();

        for (int kc = 0; kc < KS; kc += KT) {
            // ---- stage B chunk (f32 -> bf16, transposed to n-major) ----
            const float* bp = bcol + (size_t)(ks + kc) * H_;
            #pragma unroll
            for (int k8 = 0; k8 < KT; k8 += 8) {
                v8bf pb;
                #pragma unroll
                for (int j = 0; j < 8; j += 2) {
                    v2bf tb = f2bf2(bp[(size_t)(k8 + j) * H_],
                                    bp[(size_t)(k8 + j + 1) * H_]);
                    pb[j] = tb[0]; pb[j + 1] = tb[1];
                }
                *(v8bf*)&b_lds[tid * KT + k8] = pb;
            }
            if (ks + kc + KT < I_) __builtin_prefetch(bp + (size_t)KT * H_, 0, 1);
            __syncthreads();

            v16bf afr[2];
            #pragma unroll
            for (int mi = 0; mi < 2; ++mi)
                afr[mi] = load_frag(&a_slab[(mi * 16 + lrow) * KS + kc + half * 8]);
            #pragma unroll
            for (int ni = 0; ni < 2; ++ni) {
                const int n = w * 32 + ni * 16 + lrow;
                v16bf bf = load_frag(&b_lds[n * KT + half * 8]);
                #pragma unroll
                for (int mi = 0; mi < 2; ++mi)
                    acc[mi][ni] = __builtin_amdgcn_wmma_f32_16x16x32_bf16(
                        false, afr[mi], false, bf, (short)0, acc[mi][ni], false, false);
            }
            __syncthreads();
        }
    }

    // ---- epilogue: scale by routing weight, atomic accumulate into out ----
    #pragma unroll
    for (int mi = 0; mi < 2; ++mi)
        #pragma unroll
        for (int v = 0; v < 8; ++v) {
            const int ml = mi * 16 + half * 8 + v;
            if (ml < rows) {
                const int   tok2 = listTok[e * TK_ + r0 + ml];
                const float wgt  = listW[e * TK_ + r0 + ml];
                #pragma unroll
                for (int ni = 0; ni < 2; ++ni) {
                    const int nl = w * 32 + ni * 16 + lrow;
                    unsafeAtomicAdd(&out[(size_t)tok2 * H_ + nBase + nl],
                                    wgt * acc[mi][ni][v]);
                }
            }
        }
}

// ---------------------------------------------------------------------------
// Host launcher
// ---------------------------------------------------------------------------
extern "C" void kernel_launch(void* const* d_in, const int* in_sizes, int n_in,
                              void* d_out, int out_size, void* d_ws, size_t ws_size,
                              hipStream_t stream) {
    (void)in_sizes; (void)n_in; (void)out_size;
    if (ws_size < WS_NEEDED) return;   // deterministic no-op if scratch too small

    const float* x    = (const float*)d_in[0];   // (T, H)
    const float* gup  = (const float*)d_in[1];   // (E, H, 2I)
    const float* dwn  = (const float*)d_in[2];   // (E, I, H)
    const int*   idx  = (const int*)d_in[3];     // (T, K)
    const float* wts  = (const float*)d_in[4];   // (T, K)
    float*       out  = (float*)d_out;           // (T, H)

    char* ws = (char*)d_ws;
    int*    cnt     = (int*)(ws + OFF_CNT);
    int*    nTiles  = (int*)(ws + OFF_NT);
    int*    baseE   = (int*)(ws + OFF_BASE);
    int*    tileE   = (int*)(ws + OFF_TE);
    int*    tileR   = (int*)(ws + OFF_TR);
    int*    listTok = (int*)(ws + OFF_TOK);
    float*  listW   = (float*)(ws + OFF_W);
    __bf16* act     = (__bf16*)(ws + OFF_ACT);

    // 0) zero output (harness poisons it)
    k_zero<<<(T_ * H_ / 4) / 256, 256, 0, stream>>>(out);
    // 1) routing
    k_init<<<1, 32, 0, stream>>>(cnt, nTiles);
    k_route<<<T_ / 256, 256, 0, stream>>>(idx, wts, cnt, listTok, listW);
    k_tiles<<<1, 1, 0, stream>>>(cnt, baseE, tileE, tileR, nTiles);
    // 2) gate/up GEMM + silu*u  -> act (bf16)
    k_gemm1<<<dim3(MAX_TILES, I_ / NT), 256, 0, stream>>>(
        x, gup, tileE, tileR, nTiles, cnt, baseE, listTok, act);
    // 3) down GEMM + weighted scatter-add -> out (TDM-staged A panel)
    k_gemm2<<<dim3(MAX_TILES, H_ / NT), 256, G2_LDS, stream>>>(
        act, dwn, tileE, tileR, nTiles, cnt, baseE, listTok, listW, out);
}
